// SegmentVAE_11235634446657
// MI455X (gfx1250) — matrix operand order repaired and checked by reading, not compiled, not run on hardware
//
#include <hip/hip_runtime.h>
#include <math.h>

// ---------------- problem constants ----------------
#define N_SEG    16384
#define SEG_SIZE 32
#define N_ROWS   (N_SEG * SEG_SIZE)   // 524288
#define IN_DIM   128
#define HID      256
#define LAT      64
#define BN_EPS   1e-5f
#define INV_NSEG (1.0f / 16384.0f)

typedef __attribute__((ext_vector_type(16))) _Float16 v16h;
typedef __attribute__((ext_vector_type(8)))  float    v8f;

// ---------------- WMMA fragment helpers (ISA 7.12.2 layouts) ----------------
// A fragment: 16x32 f16 tile from row-major LDS (pitch in halves).
// Per-lane data = two contiguous 16B blocks: halves [k0+8*hi .. +7] and
// [k0+16+8*hi .. +7]  ->  two ds_read_b128.
__device__ inline v16h load_a_lds(const _Float16* __restrict__ A, int pitch,
                                  int m0, int k0, int lane) {
    int m  = lane & 15;
    int hi = lane >> 4;
    const _Float16* row = A + (m0 + m) * pitch + k0 + 8 * hi;
    v16h a;
    ((uint4*)&a)[0] = *(const uint4*)(row);
    ((uint4*)&a)[1] = *(const uint4*)(row + 16);
    return a;
}

// B fragment from PRE-PACKED weights: for tile (nt, kc) each lane's 16 halves
// (K = 8*hi+0..7 and 16+8*hi+0..7 of its column) are stored contiguously.
// -> two global_load_b128 per fragment, zero packing ops.
__device__ inline v16h load_b_pk(const _Float16* __restrict__ Wp, int kChunks,
                                 int nt, int kc, int lane) {
    const _Float16* p = Wp + (((size_t)nt * kChunks + kc) * 32 + lane) * 16;
    v16h b;
    ((uint4*)&b)[0] = *(const uint4*)(p);
    ((uint4*)&b)[1] = *(const uint4*)(p + 8);
    return b;
}

#define WMMA_F16(a, b, c) \
    __builtin_amdgcn_wmma_f32_16x16x32_f16(false, (a), false, (b), (short)0, (c), false, false)

// ---------------- weight pre-pack: f32 row-major [KxN] -> f16 fragment-major ----
// out[((nt*kChunks + kc)*32 + lane)*16 + j] = W[(kc*32 + (j>>3)*16 + (lane>>4)*8 + (j&7))*N + nt*16 + (lane&15)]
__global__ void packB_k(const float* __restrict__ W, _Float16* __restrict__ out,
                        int K, int N) {
    int i = blockIdx.x * 256 + threadIdx.x;
    if (i >= K * N) return;
    int j       = i & 15;
    int lane    = (i >> 4) & 31;
    int chunk   = i >> 9;            // nt*kChunks + kc
    int kChunks = K >> 5;
    int nt = chunk / kChunks;
    int kc = chunk % kChunks;
    int k = kc * 32 + ((j >> 3) * 16) + ((lane >> 4) * 8) + (j & 7);
    int n = nt * 16 + (lane & 15);
    out[i] = (_Float16)W[(size_t)k * N + n];
}

// ---------------- misc small kernels ----------------
__global__ void zero_k(float* p, int n) {
    int i = blockIdx.x * 256 + threadIdx.x;
    if (i < n) p[i] = 0.0f;
}

// column sum / sumsq over 16384 rows x 256 cols (BatchNorm batch stats)
__global__ void colstats_k(const float* __restrict__ A, float* __restrict__ sum,
                           float* __restrict__ sq) {
    int c  = threadIdx.x;          // 256 threads = 256 columns
    int r0 = blockIdx.x * 128;     // 128 blocks x 128 rows
    float s = 0.f, q = 0.f;
#pragma unroll 4
    for (int r = 0; r < 128; ++r) {
        float v = A[(size_t)(r0 + r) * HID + c];
        s += v;
        q += v * v;
    }
    atomicAdd(&sum[c], s);
    atomicAdd(&sq[c], q);
}

// out = T + bn(S)
__global__ void addbn_k(const float* __restrict__ T, const float* __restrict__ S,
                        const float* __restrict__ sum, const float* __restrict__ sq,
                        const float* __restrict__ g, const float* __restrict__ be,
                        float* __restrict__ out, int n) {
    int i = blockIdx.x * 256 + threadIdx.x;
    if (i >= n) return;
    int c    = i & (HID - 1);
    float mu = sum[c] * INV_NSEG;
    float vr = sq[c] * INV_NSEG - mu * mu;
    out[i]   = T[i] + (S[i] - mu) * rsqrtf(vr + BN_EPS) * g[c] + be[c];
}

__global__ void add_k(float* __restrict__ out, const float* __restrict__ b, int n) {
    int i = blockIdx.x * 256 + threadIdx.x;
    if (i < n) out[i] = out[i] + b[i];
}

// z = mean + eps * exp(0.5*logvar)
__global__ void reparam_k(const float* __restrict__ mean, const float* __restrict__ logvar,
                          const float* __restrict__ eps, float* __restrict__ z, int n) {
    int i = blockIdx.x * 256 + threadIdx.x;
    if (i < n) z[i] = mean[i] + eps[i] * expf(0.5f * logvar[i]);
}

// ---------------- fused encoder MLP + segment sum ----------------
// block = 256 threads (8 waves); owns 64 rows = 2 full segments.
__global__ __launch_bounds__(256)
void enc_segsum_k(const float* __restrict__ X, const _Float16* __restrict__ W1p,
                  const float* __restrict__ b1, const _Float16* __restrict__ W2p,
                  const float* __restrict__ b2, float* __restrict__ segOut) {
    __shared__ __align__(16) _Float16 sX[64 * IN_DIM];   // 16 KB
    __shared__ __align__(16) _Float16 sH[64 * HID];      // 32 KB
    __shared__ float sSeg[2 * HID];                      //  2 KB

    const int tid = threadIdx.x;
    const size_t m0g = (size_t)blockIdx.x * 64;

    // stage X tile (fp32 -> f16), coalesced
    for (int i = tid; i < 64 * IN_DIM; i += 256)
        sX[i] = (_Float16)X[m0g * IN_DIM + i];
    for (int i = tid; i < 2 * HID; i += 256) sSeg[i] = 0.0f;
    __syncthreads();

    const int lane = tid & 31;
    const int wave = tid >> 5;
    const int hi   = lane >> 4;
    const int nn   = lane & 15;

    // ---- GEMM1: H1 = relu(X[64x128] @ W1[128x256] + b1), 4x16 = 64 tiles ----
    for (int t = wave; t < 64; t += 8) {
        int mt = t >> 4, nt = t & 15;
        v8f acc = {};
#pragma unroll
        for (int kc = 0; kc < IN_DIM / 32; ++kc) {
            v16h a = load_a_lds(sX, IN_DIM, mt * 16, kc * 32, lane);
            v16h b = load_b_pk(W1p, IN_DIM / 32, nt, kc, lane);
            acc = WMMA_F16(a, b, acc);
        }
        int col  = nt * 16 + nn;
        float bv = b1[col];
#pragma unroll
        for (int r = 0; r < 8; ++r) {
            float v = fmaxf(acc[r] + bv, 0.0f);
            sH[(mt * 16 + r + 8 * hi) * HID + col] = (_Float16)v;
        }
    }
    __syncthreads();

    // ---- GEMM2: H2 = relu(H1[64x256] @ W2[256x256] + b2), + column sums ----
    for (int t = wave; t < 64; t += 8) {
        int mt = t >> 4, nt = t & 15;
        v8f acc = {};
#pragma unroll
        for (int kc = 0; kc < HID / 32; ++kc) {
            v16h a = load_a_lds(sH, HID, mt * 16, kc * 32, lane);
            v16h b = load_b_pk(W2p, HID / 32, nt, kc, lane);
            acc = WMMA_F16(a, b, acc);
        }
        int col  = nt * 16 + nn;
        float bv = b2[col];
        float ps = 0.0f;
#pragma unroll
        for (int r = 0; r < 8; ++r) ps += fmaxf(acc[r] + bv, 0.0f);
        // lanes L and L+16 hold the same column, different 8-row halves
        ps += __shfl_xor(ps, 16, 32);
        if (hi == 0) atomicAdd(&sSeg[(mt >> 1) * HID + col], ps);
    }
    __syncthreads();

    for (int i = tid; i < 2 * HID; i += 256)
        segOut[(size_t)blockIdx.x * (2 * HID) + i] = sSeg[i];
}

// ---------------- generic WMMA GEMM: out = post( pre(A)[16384xK] @ W[KxN] + bias )
// preMode: 0 = identity, 1 = BN, 2 = BN + ReLU (batch stats from csum/csq)
__global__ __launch_bounds__(256)
void gemm_k(const float* __restrict__ A, int K, const _Float16* __restrict__ Wp, int N,
            const float* __restrict__ bias, float* __restrict__ out, int preMode,
            const float* __restrict__ gamma, const float* __restrict__ beta,
            const float* __restrict__ csum, const float* __restrict__ csq, int postRelu) {
    __shared__ __align__(16) _Float16 sA[64 * HID];   // 32 KB max (K<=256)

    const int tid = threadIdx.x;
    const size_t m0g = (size_t)blockIdx.x * 64;

    for (int i = tid; i < 64 * K; i += 256) {
        int c   = i % K;
        float v = A[m0g * K + i];
        if (preMode >= 1) {
            float mu = csum[c] * INV_NSEG;
            float vr = csq[c] * INV_NSEG - mu * mu;
            v = (v - mu) * rsqrtf(vr + BN_EPS) * gamma[c] + beta[c];
            if (preMode == 2) v = fmaxf(v, 0.0f);
        }
        sA[i] = (_Float16)v;
    }
    __syncthreads();

    const int lane = tid & 31;
    const int wave = tid >> 5;
    const int hi   = lane >> 4;
    const int nn   = lane & 15;
    const int nNt  = N >> 4;           // N/16 tiles across
    const int nTot = 4 * nNt;          // 4 M-tiles (64 rows)
    const int kChunks = K >> 5;

    for (int t = wave; t < nTot; t += 8) {
        int mt = t / nNt, nt = t % nNt;
        v8f acc = {};
        for (int kc = 0; kc < kChunks; ++kc) {
            v16h a = load_a_lds(sA, K, mt * 16, kc * 32, lane);
            v16h b = load_b_pk(Wp, kChunks, nt, kc, lane);
            acc = WMMA_F16(a, b, acc);
        }
        int col  = nt * 16 + nn;
        float bv = bias[col];
#pragma unroll
        for (int r = 0; r < 8; ++r) {
            float v = acc[r] + bv;
            if (postRelu) v = fmaxf(v, 0.0f);
            out[(m0g + mt * 16 + r + 8 * hi) * (size_t)N + col] = v;
        }
    }
}

// ---------------- launcher ----------------
extern "C" void kernel_launch(void* const* d_in, const int* in_sizes, int n_in,
                              void* d_out, int out_size, void* d_ws, size_t ws_size,
                              hipStream_t stream) {
    (void)in_sizes; (void)n_in; (void)out_size; (void)ws_size;

    // inputs, in setup_inputs() order
    const float* features = (const float*)d_in[1];
    const float* eps      = (const float*)d_in[2];
    const float* w_enc1 = (const float*)d_in[3];   const float* b_enc1 = (const float*)d_in[4];
    const float* w_enc2 = (const float*)d_in[5];   const float* b_enc2 = (const float*)d_in[6];
    const float* bn_g   = (const float*)d_in[7];   const float* bn_b   = (const float*)d_in[8];
    const float* w_l0   = (const float*)d_in[9];   const float* b_l0   = (const float*)d_in[10];
    const float* w_l1   = (const float*)d_in[11];  const float* b_l1   = (const float*)d_in[12];
    const float* w_mean = (const float*)d_in[13];  const float* b_mean = (const float*)d_in[14];
    const float* w_lv   = (const float*)d_in[15];  const float* b_lv   = (const float*)d_in[16];
    const float* w_d1   = (const float*)d_in[17];  const float* b_d1   = (const float*)d_in[18];
    const float* bn1_g  = (const float*)d_in[19];  const float* bn1_b  = (const float*)d_in[20];
    const float* w_d2   = (const float*)d_in[21];  const float* b_d2   = (const float*)d_in[22];
    const float* bn2_g  = (const float*)d_in[23];  const float* bn2_b  = (const float*)d_in[24];
    const float* w_d3   = (const float*)d_in[25];  const float* b_d3   = (const float*)d_in[26];

    // output slices (concatenated in return order)
    float* out    = (float*)d_out;
    float* meanO  = out;                               // 16384*64
    float* lvO    = meanO + (size_t)N_SEG * LAT;       // 16384*64
    float* zO     = lvO   + (size_t)N_SEG * LAT;       // 16384*64
    float* reconO = zO    + (size_t)N_SEG * LAT;       // 16384*256
    float* segO   = reconO + (size_t)N_SEG * HID;      // 16384*256

    // workspace layout (all offsets 32B-aligned by construction)
    char* ws = (char*)d_ws;
    size_t o = 0;
    _Float16* W1p  = (_Float16*)(ws + o); o += (size_t)IN_DIM * HID * 2;
    _Float16* W2p  = (_Float16*)(ws + o); o += (size_t)HID * HID * 2;
    _Float16* Wl0p = (_Float16*)(ws + o); o += (size_t)HID * HID * 2;
    _Float16* Wl1p = (_Float16*)(ws + o); o += (size_t)HID * HID * 2;
    _Float16* Wmp  = (_Float16*)(ws + o); o += (size_t)HID * LAT * 2;
    _Float16* Wlvp = (_Float16*)(ws + o); o += (size_t)HID * LAT * 2;
    _Float16* Wd1p = (_Float16*)(ws + o); o += (size_t)LAT * HID * 2;
    _Float16* Wd2p = (_Float16*)(ws + o); o += (size_t)HID * HID * 2;
    _Float16* Wd3p = (_Float16*)(ws + o); o += (size_t)HID * HID * 2;
    o = (o + 255) & ~(size_t)255;
    float* stats = (float*)(ws + o); o += 6 * HID * 4;   // [sum0,sq0,sum1,sq1,sum2,sq2]
    float* sum0 = stats,           * sq0 = stats + HID;
    float* sum1 = stats + 2 * HID, * sq1 = stats + 3 * HID;
    float* sum2 = stats + 4 * HID, * sq2 = stats + 5 * HID;
    o = (o + 255) & ~(size_t)255;
    float* T  = (float*)(ws + o); o += (size_t)N_SEG * HID * 4;   // 16 MB
    float* X1 = (float*)(ws + o); o += (size_t)N_SEG * HID * 4;   // 16 MB

    const int nSegHid = N_SEG * HID;   // 4194304
    const int nSegLat = N_SEG * LAT;   // 1048576

    // 0) zero BN accumulators (re-zeroed every call -> deterministic)
    zero_k<<<(6 * HID + 255) / 256, 256, 0, stream>>>(stats, 6 * HID);

    // 1) weights fp32 -> f16, packed into per-lane WMMA B-fragment layout
    packB_k<<<(IN_DIM * HID + 255) / 256, 256, 0, stream>>>(w_enc1, W1p, IN_DIM, HID);
    packB_k<<<(HID * HID + 255) / 256, 256, 0, stream>>>(w_enc2, W2p, HID, HID);
    packB_k<<<(HID * HID + 255) / 256, 256, 0, stream>>>(w_l0, Wl0p, HID, HID);
    packB_k<<<(HID * HID + 255) / 256, 256, 0, stream>>>(w_l1, Wl1p, HID, HID);
    packB_k<<<(HID * LAT + 255) / 256, 256, 0, stream>>>(w_mean, Wmp, HID, LAT);
    packB_k<<<(HID * LAT + 255) / 256, 256, 0, stream>>>(w_lv, Wlvp, HID, LAT);
    packB_k<<<(LAT * HID + 255) / 256, 256, 0, stream>>>(w_d1, Wd1p, LAT, HID);
    packB_k<<<(HID * HID + 255) / 256, 256, 0, stream>>>(w_d2, Wd2p, HID, HID);
    packB_k<<<(HID * HID + 255) / 256, 256, 0, stream>>>(w_d3, Wd3p, HID, HID);

    // 2) fused encoder MLP + segment sum  (the 103-GFLOP hot loop)
    enc_segsum_k<<<N_ROWS / 64, 256, 0, stream>>>(features, W1p, b_enc1, W2p, b_enc2, segO);

    // 3) BN0 batch stats over seg_sum
    colstats_k<<<N_SEG / 128, 256, 0, stream>>>(segO, sum0, sq0);

    // 4) residual block 0: T = relu(bn(seg)@w_l0+b);  X1 = T + bn(seg)
    gemm_k<<<N_SEG / 64, 256, 0, stream>>>(segO, HID, Wl0p, HID, b_l0, T, 1,
                                           bn_g, bn_b, sum0, sq0, 1);
    addbn_k<<<(nSegHid + 255) / 256, 256, 0, stream>>>(T, segO, sum0, sq0, bn_g, bn_b,
                                                       X1, nSegHid);

    // 5) residual block 1: T = relu(X1@w_l1+b);  X1 += T
    gemm_k<<<N_SEG / 64, 256, 0, stream>>>(X1, HID, Wl1p, HID, b_l1, T, 0,
                                           nullptr, nullptr, nullptr, nullptr, 1);
    add_k<<<(nSegHid + 255) / 256, 256, 0, stream>>>(X1, T, nSegHid);

    // 6) heads + reparameterization
    gemm_k<<<N_SEG / 64, 256, 0, stream>>>(X1, HID, Wmp, LAT, b_mean, meanO, 0,
                                           nullptr, nullptr, nullptr, nullptr, 0);
    gemm_k<<<N_SEG / 64, 256, 0, stream>>>(X1, HID, Wlvp, LAT, b_lv, lvO, 0,
                                           nullptr, nullptr, nullptr, nullptr, 0);
    reparam_k<<<(nSegLat + 255) / 256, 256, 0, stream>>>(meanO, lvO, eps, zO, nSegLat);

    // 7) decoder
    gemm_k<<<N_SEG / 64, 256, 0, stream>>>(zO, LAT, Wd1p, HID, b_d1, T, 0,
                                           nullptr, nullptr, nullptr, nullptr, 0);
    colstats_k<<<N_SEG / 128, 256, 0, stream>>>(T, sum1, sq1);
    gemm_k<<<N_SEG / 64, 256, 0, stream>>>(T, HID, Wd2p, HID, b_d2, X1, 2,
                                           bn1_g, bn1_b, sum1, sq1, 0);
    colstats_k<<<N_SEG / 128, 256, 0, stream>>>(X1, sum2, sq2);
    gemm_k<<<N_SEG / 64, 256, 0, stream>>>(X1, HID, Wd3p, HID, b_d3, reconO, 2,
                                           bn2_g, bn2_b, sum2, sq2, 0);
}
